// MultiHeadAttention_1245540516063
// MI455X (gfx1250) — matrix operand order repaired and checked by reading, not compiled, not run on hardware
//
#include <hip/hip_runtime.h>
#include <hip/hip_bf16.h>

#define DM    768
#define SEQ   1024
#define NH    12
#define BATCH 8

#define AS1 __attribute__((address_space(1)))
#define AS3 __attribute__((address_space(3)))

typedef __attribute__((ext_vector_type(16))) __bf16          v16bf;
typedef __attribute__((ext_vector_type(2)))  __bf16          bf16x2;
typedef __attribute__((ext_vector_type(8)))  float           v8f;
typedef __attribute__((ext_vector_type(2)))  float           f32x2;
typedef __attribute__((ext_vector_type(8)))  unsigned short  u16x8;
typedef __attribute__((ext_vector_type(4)))  unsigned short  u16x4;
typedef __attribute__((ext_vector_type(4)))  float           f32x4;
typedef __attribute__((ext_vector_type(4)))  int             i32x4;

#if defined(__has_builtin)
#  if __has_builtin(__builtin_amdgcn_cvt_pk_bf16_f32)
#    define HAS_CVT_PK_BF16 1
#  endif
#  if __has_builtin(__builtin_amdgcn_global_load_async_to_lds_b128)
#    define HAS_ASYNC_LDS 1
#  endif
#  if __has_builtin(__builtin_amdgcn_s_wait_asynccnt)
#    define HAS_WAIT_ASYNC 1
#  endif
#endif

union Frag  { u16x8 h[2]; v16bf v; };
union Pack4 { unsigned u[2]; u16x4 v; };
union Pack8 { unsigned u[4]; u16x8 v; };

// pack two f32 -> {lo=a, hi=b} bf16 pair (native converts, RTNE)
__device__ __forceinline__ unsigned pk_bf16(float a, float b) {
#if defined(HAS_CVT_PK_BF16)
    return __builtin_bit_cast(unsigned, __builtin_amdgcn_cvt_pk_bf16_f32(a, b));
#else
    f32x2 t; t[0] = a; t[1] = b;
    bf16x2 r = __builtin_convertvector(t, bf16x2);
    return __builtin_bit_cast(unsigned, r);
#endif
}

__device__ __forceinline__ v8f wmma_bf16(const Frag& a, const Frag& b, v8f c) {
    return __builtin_amdgcn_wmma_f32_16x16x32_bf16(false, a.v, false, b.v,
                                                   (short)0, c, false, false);
}

__device__ __forceinline__ v8f vzero8() {
    v8f z;
#pragma unroll
    for (int i = 0; i < 8; i++) z[i] = 0.0f;
    return z;
}

__device__ __forceinline__ void wait_async_then_barrier() {
#if defined(HAS_ASYNC_LDS)
#  if defined(HAS_WAIT_ASYNC)
    __builtin_amdgcn_s_wait_asynccnt(0);
#  else
    asm volatile("s_wait_asynccnt 0" ::: "memory");
#  endif
#endif
    __syncthreads();
}

// ---------------------------------------------------------------------------
// K1: projection GEMM  dst[b,h,s,d] (bf16) = (A[8192,768] @ W[768,768] + bias)
// Block tile 128x128, K-step 32, 8 waves in 4(m) x 2(n); wave tile 32x64.
// ---------------------------------------------------------------------------
__global__ __launch_bounds__(256) void k_proj(const float* __restrict__ A,
                                              const float* __restrict__ W,
                                              const float* __restrict__ bias,
                                              unsigned short* __restrict__ dst) {
    __shared__ unsigned short As[128 * 32];   // [row][k]  bf16
    __shared__ unsigned short Bs[128 * 32];   // [col][k]  bf16 (W transposed)
    const int tid = threadIdx.x;
    const int w = tid >> 5, lane = tid & 31, lo = lane & 15, hi = lane >> 4;
    const int wm = w & 3, wn = w >> 2;
    const int rowBase = blockIdx.y * 128;
    const int colBase = blockIdx.x * 128;

    v8f acc[2][4];
#pragma unroll
    for (int i = 0; i < 2; i++)
#pragma unroll
        for (int j = 0; j < 4; j++) acc[i][j] = vzero8();

    for (int kk = 0; kk < DM; kk += 32) {
        if (kk + 32 < DM) {   // prefetch next K tile (global_prefetch_b8)
            __builtin_prefetch(&A[(size_t)(rowBase + (tid & 127)) * DM + kk + 32], 0, 3);
            __builtin_prefetch(&W[(size_t)(kk + 32 + (tid & 31)) * DM + colBase], 0, 3);
        }
        // stage A tile 128x32 (f32 -> bf16)
#pragma unroll
        for (int i = 0; i < 4; i++) {
            int idx = tid + i * 256;                 // 1024 float4's
            int r = idx >> 3, c4 = idx & 7;
            f32x4 v = *(const f32x4*)&A[(size_t)(rowBase + r) * DM + kk + c4 * 4];
            Pack4 o;
            o.u[0] = pk_bf16(v[0], v[1]);
            o.u[1] = pk_bf16(v[2], v[3]);
            *(u16x4*)&As[r * 32 + c4 * 4] = o.v;
        }
        // stage W tile 32x128 transposed -> Bs[col][k]
#pragma unroll
        for (int i = 0; i < 4; i++) {
            int idx = tid + i * 256;
            int r = idx >> 5, c4 = idx & 31;         // r = k, c4 = col/4
            f32x4 v = *(const f32x4*)&W[(size_t)(kk + r) * DM + colBase + c4 * 4];
            unsigned p0 = pk_bf16(v[0], v[1]);
            unsigned p1 = pk_bf16(v[2], v[3]);
            Bs[(c4 * 4 + 0) * 32 + r] = (unsigned short)p0;
            Bs[(c4 * 4 + 1) * 32 + r] = (unsigned short)(p0 >> 16);
            Bs[(c4 * 4 + 2) * 32 + r] = (unsigned short)p1;
            Bs[(c4 * 4 + 3) * 32 + r] = (unsigned short)(p1 >> 16);
        }
        __syncthreads();

        Frag a0, a1;
        {
            int r0 = wm * 32 + lo;
            a0.h[0] = *(const u16x8*)&As[r0 * 32 + hi * 8];
            a0.h[1] = *(const u16x8*)&As[r0 * 32 + 16 + hi * 8];
            int r1 = wm * 32 + 16 + lo;
            a1.h[0] = *(const u16x8*)&As[r1 * 32 + hi * 8];
            a1.h[1] = *(const u16x8*)&As[r1 * 32 + 16 + hi * 8];
        }
#pragma unroll
        for (int nt = 0; nt < 4; nt++) {
            Frag bf;
            int c = wn * 64 + nt * 16 + lo;
            bf.h[0] = *(const u16x8*)&Bs[c * 32 + hi * 8];
            bf.h[1] = *(const u16x8*)&Bs[c * 32 + 16 + hi * 8];
            acc[0][nt] = wmma_bf16(a0, bf, acc[0][nt]);
            acc[1][nt] = wmma_bf16(a1, bf, acc[1][nt]);
        }
        __syncthreads();
    }
    // epilogue: +bias, bf16, remap to [B,H,S,64]
#pragma unroll
    for (int mt = 0; mt < 2; mt++)
#pragma unroll
        for (int nt = 0; nt < 4; nt++) {
            int j = colBase + wn * 64 + nt * 16 + lo;
            float bj = bias[j];
            int h = j >> 6, d = j & 63;
#pragma unroll
            for (int r = 0; r < 8; r += 2) {
                int m = rowBase + wm * 32 + mt * 16 + r + hi * 8;
                int b = m >> 10, s = m & 1023;
                unsigned p = pk_bf16(acc[mt][nt][r] + bj, acc[mt][nt][r + 1] + bj);
                size_t base = (((size_t)(b * NH + h)) * SEQ + s) * 64 + d;
                dst[base]      = (unsigned short)p;         // row s
                dst[base + 64] = (unsigned short)(p >> 16); // row s+1
            }
        }
}

// ---------------------------------------------------------------------------
// K2: scores = (q @ k^T) * 1/8, mask-fill.  Fragments loaded from global.
// ---------------------------------------------------------------------------
__global__ __launch_bounds__(256) void k_scores(const unsigned short* __restrict__ qh,
                                                const unsigned short* __restrict__ kh,
                                                const unsigned char* __restrict__ mask,
                                                float* __restrict__ attn) {
    const int tid = threadIdx.x;
    const int w = tid >> 5, lane = tid & 31, lo = lane & 15, hi = lane >> 4;
    const int bh = blockIdx.y, b = bh / NH;
    const int qbase = blockIdx.x * 128;
    const int qrow_lane = qbase + w * 16 + lo;
    const size_t qoff = ((size_t)bh * SEQ + qrow_lane) * 64;

    Frag a[2];
#pragma unroll
    for (int ks = 0; ks < 2; ks++) {
        a[ks].h[0] = *(const u16x8*)&qh[qoff + ks * 32 + hi * 8];
        a[ks].h[1] = *(const u16x8*)&qh[qoff + ks * 32 + 16 + hi * 8];
    }

    for (int nc = 0; nc < SEQ; nc += 128) {
        v8f acc[8];
#pragma unroll
        for (int nt = 0; nt < 8; nt++) acc[nt] = vzero8();
#pragma unroll
        for (int ks = 0; ks < 2; ks++) {
#pragma unroll
            for (int nt = 0; nt < 8; nt++) {
                int key = nc + nt * 16 + lo;
                size_t koff = ((size_t)bh * SEQ + key) * 64 + ks * 32 + hi * 8;
                Frag bf;
                bf.h[0] = *(const u16x8*)&kh[koff];
                bf.h[1] = *(const u16x8*)&kh[koff + 16];
                acc[nt] = wmma_bf16(a[ks], bf, acc[nt]);
            }
        }
        const float scale = 0.125f;   // 1/sqrt(64)
#pragma unroll
        for (int nt = 0; nt < 8; nt++)
#pragma unroll
            for (int r = 0; r < 8; r++) {
                int qr = qbase + w * 16 + r + hi * 8;
                int col = nc + nt * 16 + lo;
                float v = acc[nt][r] * scale;
                if (mask[((size_t)b * SEQ + qr) * SEQ + col]) v = -1e9f;
                attn[((size_t)bh * SEQ + qr) * SEQ + col] = v;
            }
    }
}

// ---------------------------------------------------------------------------
// K3: row softmax in place; one wave per 1024-wide row.
// ---------------------------------------------------------------------------
__global__ __launch_bounds__(256) void k_softmax(float* __restrict__ attn) {
    const int tid = threadIdx.x, w = tid >> 5, lane = tid & 31;
    const size_t row = (size_t)blockIdx.x * 8 + w;
    float* p = attn + row * SEQ;
    float v[32];
#pragma unroll
    for (int j = 0; j < 8; j++) {
        f32x4 x = *(const f32x4*)&p[j * 128 + lane * 4];
        v[j * 4 + 0] = x[0]; v[j * 4 + 1] = x[1]; v[j * 4 + 2] = x[2]; v[j * 4 + 3] = x[3];
    }
    float m = -3.4e38f;
#pragma unroll
    for (int i = 0; i < 32; i++) m = fmaxf(m, v[i]);
#pragma unroll
    for (int off = 16; off; off >>= 1) m = fmaxf(m, __shfl_xor(m, off, 32));
    float s = 0.0f;
#pragma unroll
    for (int i = 0; i < 32; i++) { v[i] = __expf(v[i] - m); s += v[i]; }
#pragma unroll
    for (int off = 16; off; off >>= 1) s += __shfl_xor(s, off, 32);
    float inv = 1.0f / s;
#pragma unroll
    for (int j = 0; j < 8; j++) {
        f32x4 x;
        x[0] = v[j * 4 + 0] * inv; x[1] = v[j * 4 + 1] * inv;
        x[2] = v[j * 4 + 2] * inv; x[3] = v[j * 4 + 3] * inv;
        *(f32x4*)&p[j * 128 + lane * 4] = x;
    }
}

// ---------------------------------------------------------------------------
// K4: context = attn @ v  (M=1024, N=64, K=1024 per head).
// ---------------------------------------------------------------------------
__global__ __launch_bounds__(256) void k_pv(const float* __restrict__ attn,
                                            const unsigned short* __restrict__ vh,
                                            unsigned short* __restrict__ ctx) {
    __shared__ unsigned short vhT[64 * 128];   // [n][k] bf16, 16 KB
    const int tid = threadIdx.x;
    const int w = tid >> 5, lane = tid & 31, lo = lane & 15, hi = lane >> 4;
    const int bh = blockIdx.y, b = bh / NH, h = bh % NH;
    const int qbase = blockIdx.x * 128;

    v8f acc[4];
#pragma unroll
    for (int nt = 0; nt < 4; nt++) acc[nt] = vzero8();

    for (int kc = 0; kc < 8; kc++) {              // 128 keys per chunk
#pragma unroll
        for (int i = 0; i < 8; i++) {
            int idx = tid + i * 256;              // 2048 u16x4's
            int r = idx >> 4, c4 = idx & 15;
            u16x4 x = *(const u16x4*)&vh[((size_t)bh * SEQ + kc * 128 + r) * 64 + c4 * 4];
#pragma unroll
            for (int j = 0; j < 4; j++) vhT[(c4 * 4 + j) * 128 + r] = x[j];
        }
        __syncthreads();
#pragma unroll
        for (int ks = 0; ks < 4; ks++) {          // K-steps of 32
            int qr = qbase + w * 16 + lo;
            size_t abase = ((size_t)bh * SEQ + qr) * SEQ + kc * 128 + ks * 32;
            f32x4 f0 = *(const f32x4*)&attn[abase + hi * 8];
            f32x4 f1 = *(const f32x4*)&attn[abase + hi * 8 + 4];
            f32x4 f2 = *(const f32x4*)&attn[abase + 16 + hi * 8];
            f32x4 f3 = *(const f32x4*)&attn[abase + 16 + hi * 8 + 4];
            Frag a;
            Pack8 p0, p1;
            p0.u[0] = pk_bf16(f0[0], f0[1]);
            p0.u[1] = pk_bf16(f0[2], f0[3]);
            p0.u[2] = pk_bf16(f1[0], f1[1]);
            p0.u[3] = pk_bf16(f1[2], f1[3]);
            p1.u[0] = pk_bf16(f2[0], f2[1]);
            p1.u[1] = pk_bf16(f2[2], f2[3]);
            p1.u[2] = pk_bf16(f3[0], f3[1]);
            p1.u[3] = pk_bf16(f3[2], f3[3]);
            a.h[0] = p0.v; a.h[1] = p1.v;
#pragma unroll
            for (int nt = 0; nt < 4; nt++) {
                Frag bf;
                int c = nt * 16 + lo;
                bf.h[0] = *(const u16x8*)&vhT[c * 128 + ks * 32 + hi * 8];
                bf.h[1] = *(const u16x8*)&vhT[c * 128 + ks * 32 + 16 + hi * 8];
                acc[nt] = wmma_bf16(a, bf, acc[nt]);
            }
        }
        __syncthreads();
    }
#pragma unroll
    for (int nt = 0; nt < 4; nt++) {
        int d = nt * 16 + lo;
#pragma unroll
        for (int r = 0; r < 8; r += 2) {
            int m = qbase + w * 16 + r + hi * 8;
            unsigned p = pk_bf16(acc[nt][r], acc[nt][r + 1]);
            size_t base = ((size_t)(b * SEQ + m)) * DM + h * 64 + d;
            ctx[base]      = (unsigned short)p;
            ctx[base + DM] = (unsigned short)(p >> 16);
        }
    }
}

// ---------------------------------------------------------------------------
// K5: out_pre = ctx(bf16) @ W_O + b_O + residual  (f32)
// A tile staged via async global->LDS DMA when available.
// ---------------------------------------------------------------------------
__global__ __launch_bounds__(256) void k_oproj(const unsigned short* __restrict__ Actx,
                                               const float* __restrict__ W,
                                               const float* __restrict__ bias,
                                               const float* __restrict__ resid,
                                               float* __restrict__ out) {
    __shared__ unsigned short As[128 * 32];
    __shared__ unsigned short Bs[128 * 32];
    const int tid = threadIdx.x;
    const int w = tid >> 5, lane = tid & 31, lo = lane & 15, hi = lane >> 4;
    const int wm = w & 3, wn = w >> 2;
    const int rowBase = blockIdx.y * 128;
    const int colBase = blockIdx.x * 128;

    v8f acc[2][4];
#pragma unroll
    for (int i = 0; i < 2; i++)
#pragma unroll
        for (int j = 0; j < 4; j++) acc[i][j] = vzero8();

    for (int kk = 0; kk < DM; kk += 32) {
        if (kk + 32 < DM) {
            __builtin_prefetch(&Actx[(size_t)(rowBase + (tid & 127)) * DM + kk + 32], 0, 3);
            __builtin_prefetch(&W[(size_t)(kk + 32 + (tid & 31)) * DM + colBase], 0, 3);
        }
        // stage A tile (bf16, no conversion): async DMA to LDS if available
#if defined(HAS_ASYNC_LDS)
#pragma unroll
        for (int i = 0; i < 2; i++) {
            int idx = tid + i * 256;             // 512 b128 transfers
            int r = idx >> 2, c8 = idx & 3;
            const unsigned short* g = &Actx[(size_t)(rowBase + r) * DM + kk + c8 * 8];
            __builtin_amdgcn_global_load_async_to_lds_b128(
                (AS1 i32x4*)g,
                (AS3 i32x4*)&As[r * 32 + c8 * 8],
                0, 0);
        }
#else
#pragma unroll
        for (int i = 0; i < 2; i++) {
            int idx = tid + i * 256;
            int r = idx >> 2, c8 = idx & 3;
            u16x8 x = *(const u16x8*)&Actx[(size_t)(rowBase + r) * DM + kk + c8 * 8];
            *(u16x8*)&As[r * 32 + c8 * 8] = x;
        }
#endif
        // stage W tile transposed (f32 -> bf16)
#pragma unroll
        for (int i = 0; i < 4; i++) {
            int idx = tid + i * 256;
            int r = idx >> 5, c4 = idx & 31;
            f32x4 v = *(const f32x4*)&W[(size_t)(kk + r) * DM + colBase + c4 * 4];
            unsigned p0 = pk_bf16(v[0], v[1]);
            unsigned p1 = pk_bf16(v[2], v[3]);
            Bs[(c4 * 4 + 0) * 32 + r] = (unsigned short)p0;
            Bs[(c4 * 4 + 1) * 32 + r] = (unsigned short)(p0 >> 16);
            Bs[(c4 * 4 + 2) * 32 + r] = (unsigned short)p1;
            Bs[(c4 * 4 + 3) * 32 + r] = (unsigned short)(p1 >> 16);
        }
        wait_async_then_barrier();

        Frag a0, a1;
        {
            int r0 = wm * 32 + lo;
            a0.h[0] = *(const u16x8*)&As[r0 * 32 + hi * 8];
            a0.h[1] = *(const u16x8*)&As[r0 * 32 + 16 + hi * 8];
            int r1 = wm * 32 + 16 + lo;
            a1.h[0] = *(const u16x8*)&As[r1 * 32 + hi * 8];
            a1.h[1] = *(const u16x8*)&As[r1 * 32 + 16 + hi * 8];
        }
#pragma unroll
        for (int nt = 0; nt < 4; nt++) {
            Frag bf;
            int c = wn * 64 + nt * 16 + lo;
            bf.h[0] = *(const u16x8*)&Bs[c * 32 + hi * 8];
            bf.h[1] = *(const u16x8*)&Bs[c * 32 + 16 + hi * 8];
            acc[0][nt] = wmma_bf16(a0, bf, acc[0][nt]);
            acc[1][nt] = wmma_bf16(a1, bf, acc[1][nt]);
        }
        __syncthreads();
    }
#pragma unroll
    for (int mt = 0; mt < 2; mt++)
#pragma unroll
        for (int nt = 0; nt < 4; nt++)
#pragma unroll
            for (int r = 0; r < 8; r++) {
                int m = rowBase + wm * 32 + mt * 16 + r + hi * 8;
                int j = colBase + wn * 64 + nt * 16 + lo;
                out[(size_t)m * DM + j] =
                    acc[mt][nt][r] + bias[j] + resid[(size_t)m * DM + j];
            }
}

// ---------------------------------------------------------------------------
// K6: LayerNorm over last dim (768); one wave per row.
// ---------------------------------------------------------------------------
__global__ __launch_bounds__(256) void k_ln(const float* __restrict__ x,
                                            const float* __restrict__ g,
                                            const float* __restrict__ be,
                                            float* __restrict__ out) {
    const int tid = threadIdx.x, w = tid >> 5, lane = tid & 31;
    const size_t row = (size_t)blockIdx.x * 8 + w;
    const float* p = x + row * DM;
    float v[24];
#pragma unroll
    for (int j = 0; j < 6; j++) {
        f32x4 t = *(const f32x4*)&p[j * 128 + lane * 4];
        v[j * 4 + 0] = t[0]; v[j * 4 + 1] = t[1]; v[j * 4 + 2] = t[2]; v[j * 4 + 3] = t[3];
    }
    float s = 0.0f, s2 = 0.0f;
#pragma unroll
    for (int i = 0; i < 24; i++) { s += v[i]; s2 += v[i] * v[i]; }
#pragma unroll
    for (int off = 16; off; off >>= 1) {
        s += __shfl_xor(s, off, 32);
        s2 += __shfl_xor(s2, off, 32);
    }
    float mu = s * (1.0f / DM);
    float var = s2 * (1.0f / DM) - mu * mu;
    float rs = rsqrtf(var + 1e-5f);
#pragma unroll
    for (int j = 0; j < 6; j++) {
        int col = j * 128 + lane * 4;
        f32x4 o;
#pragma unroll
        for (int q = 0; q < 4; q++)
            o[q] = (v[j * 4 + q] - mu) * rs * g[col + q] + be[col + q];
        *(f32x4*)&out[row * DM + col] = o;
    }
}

// ---------------------------------------------------------------------------
extern "C" void kernel_launch(void* const* d_in, const int* in_sizes, int n_in,
                              void* d_out, int out_size, void* d_ws, size_t ws_size,
                              hipStream_t stream) {
    const float* Q   = (const float*)d_in[0];
    const float* K   = (const float*)d_in[1];
    const float* V   = (const float*)d_in[2];
    const float* W_Q = (const float*)d_in[3];
    const float* b_Q = (const float*)d_in[4];
    const float* W_K = (const float*)d_in[5];
    const float* b_K = (const float*)d_in[6];
    const float* W_V = (const float*)d_in[7];
    const float* b_V = (const float*)d_in[8];
    const float* W_O = (const float*)d_in[9];
    const float* b_O = (const float*)d_in[10];
    const float* gam = (const float*)d_in[11];
    const float* bet = (const float*)d_in[12];
    const unsigned char* mask = (const unsigned char*)d_in[13];

    float* out  = (float*)d_out;
    float* attn = out + (size_t)BATCH * SEQ * DM;

    const size_t HEADS_ELEMS = (size_t)BATCH * NH * SEQ * 64;   // bf16 elems
    unsigned short* qh  = (unsigned short*)d_ws;
    unsigned short* kh  = qh + HEADS_ELEMS;
    unsigned short* vhp = kh + HEADS_ELEMS;
    unsigned short* ctx = vhp + HEADS_ELEMS;                // [B*S, 768] bf16
    float*          tmp = (float*)(ctx + (size_t)BATCH * SEQ * DM);

    dim3 blk(256);
    dim3 gproj(DM / 128, BATCH * SEQ / 128);                // (6, 64)
    k_proj<<<gproj, blk, 0, stream>>>(Q, W_Q, b_Q, qh);
    k_proj<<<gproj, blk, 0, stream>>>(K, W_K, b_K, kh);
    k_proj<<<gproj, blk, 0, stream>>>(V, W_V, b_V, vhp);

    dim3 gatt(SEQ / 128, BATCH * NH);                       // (8, 96)
    k_scores<<<gatt, blk, 0, stream>>>(qh, kh, mask, attn);
    k_softmax<<<dim3(BATCH * NH * SEQ / 8), blk, 0, stream>>>(attn);
    k_pv<<<gatt, blk, 0, stream>>>(attn, vhp, ctx);

    k_oproj<<<gproj, blk, 0, stream>>>(ctx, W_O, b_O, Q, tmp);
    k_ln<<<dim3(BATCH * SEQ / 8), blk, 0, stream>>>(tmp, gam, bet, out);
}
